// ProgressiveFocusedAttention_455266533868
// MI455X (gfx1250) — compile-verified, matrix operand order checked by reading, not run on hardware
//
#include <hip/hip_runtime.h>
#include <hip/hip_bf16.h>

typedef __attribute__((ext_vector_type(16))) _Float16 v16h;
typedef __attribute__((ext_vector_type(8)))  float    v8f;
typedef __attribute__((ext_vector_type(4)))  float    f4;   // native vec: OK for nontemporal builtins
typedef __attribute__((ext_vector_type(4)))  unsigned int u4;

#define DIM   192
#define NH    6
#define HD    32
#define NTOK  64
#define SHIFT 4
#define NWIN  1024
#define C3    576
#define VT_LD 72   // padded n-stride for transposed v' in LDS (bank-conflict break)

// ---- CDNA5 wave32 WMMA f16 fragment helpers (ISA 7.12.2 layouts) ----
// A/B 16-bit fragment: lanes 0-15 hold K = {0..7, 16..23}+kbase,
//                      lanes 16-31 hold K = {8..15, 24..31}+kbase.

// Contiguous (row-major along K) fragment: two 16-byte loads.
__device__ __forceinline__ v16h frag_row(const _Float16* p0, int row, int ld, int kbase) {
  int lane = threadIdx.x & 31;
  int klo  = (lane & 16) ? 8 : 0;
  const _Float16* p = p0 + row * ld + kbase + klo;
  v16h f;
#pragma unroll
  for (int t = 0; t < 8; ++t) f[t] = p[t];
#pragma unroll
  for (int t = 0; t < 8; ++t) f[8 + t] = p[16 + t];
  return f;
}

// Pre-swizzled fragment: lane's 16 halves stored contiguously (32 B / lane).
__device__ __forceinline__ v16h frag_swz(const _Float16* p0, int blk) {
  int lane = threadIdx.x & 31;
  const _Float16* p = p0 + (((size_t)blk * 32 + lane) << 4);
  v16h f;
#pragma unroll
  for (int t = 0; t < 16; ++t) f[t] = p[t];
  return f;
}

// ---- Kernel 0: weights fp32 -> fp16, stored in B-fragment order ----
__global__ __launch_bounds__(256) void k_cvt(const float* __restrict__ qkv_w,
                                             const float* __restrict__ proj_w,
                                             _Float16* __restrict__ wqkv,
                                             _Float16* __restrict__ wproj) {
  int i = blockIdx.x * 256 + threadIdx.x;
  if (i < DIM * C3) {
    int t = i & 15, lane = (i >> 4) & 31, blk = i >> 9;  // blk = jt*6+kt, NJ=36
    int kt = blk % 6, jt = blk / 6;
    int col = jt * 16 + (lane & 15);
    int klo = (lane & 16) ? 8 : 0;
    int k = kt * 32 + klo + ((t & 8) ? (8 + t) : t);
    wqkv[i] = (_Float16)qkv_w[k * C3 + col];
  } else {
    int j = i - DIM * C3;
    if (j < DIM * DIM) {
      int t = j & 15, lane = (j >> 4) & 31, blk = j >> 9; // blk = jt*6+kt, NJ=12
      int kt = blk % 6, jt = blk / 6;
      int col = jt * 16 + (lane & 15);
      int klo = (lane & 16) ? 8 : 0;
      int k = kt * 32 + klo + ((t & 8) ? (8 + t) : t);
      wproj[j] = (_Float16)proj_w[k * DIM + col];
    }
  }
}

// ------- Kernel 1: shifted-window gather + QKV GEMM + LePE conv -------
__global__ __launch_bounds__(256) void k_qkv_lepe(
    const float* __restrict__ x, const float* __restrict__ qkv_b,
    const float* __restrict__ lepe_w, const float* __restrict__ lepe_b,
    const _Float16* __restrict__ wqkv,
    _Float16* __restrict__ qf, _Float16* __restrict__ kf, _Float16* __restrict__ vf) {
  // lds_x: activations during GEMM; reused as transposed v' (padded) after
  __shared__ __align__(16) _Float16 lds_x[DIM * VT_LD];     // 27 KB
  __shared__ __align__(16) _Float16 lds_qk[2 * NTOK * DIM]; // 48 KB q|k f16
  __shared__ __align__(16) float    lds_v[NTOK * DIM];      // 48 KB v f32
  int w = blockIdx.x;
  int b = w >> 8, wh = (w >> 4) & 15, wwi = w & 15;
  int tid = threadIdx.x;

  // gather: shifted = roll(x, -SHIFT) -> shifted[i] = x[(i+SHIFT) & 127]
  for (int idx = tid; idx < NTOK * (DIM / 4); idx += 256) {
    int n = idx / (DIM / 4), c4 = idx - n * (DIM / 4);
    int r = n >> 3, c = n & 7;
    int sh = (wh * 8 + r + SHIFT) & 127;
    int sw = (wwi * 8 + c + SHIFT) & 127;
    f4 v = __builtin_nontemporal_load(
        (const f4*)(x + (((size_t)b * 128 + sh) * 128 + sw) * DIM + c4 * 4));
    _Float16* d = lds_x + n * DIM + c4 * 4;
    d[0] = (_Float16)v.x; d[1] = (_Float16)v.y;
    d[2] = (_Float16)v.z; d[3] = (_Float16)v.w;
  }
  __syncthreads();

  int lane = tid & 31, wave = tid >> 5;
  int col  = lane & 15;
  int radd = (lane & 16) ? 8 : 0;

  // 64x576 = (4 m-tiles) x (36 j-tiles), K = 192 (6 steps of 32)
  for (int t = wave; t < 144; t += 8) {
    int mt = t / 36, jt = t - mt * 36;
    int jbase = jt * 16;
    float bias = qkv_b[jbase + col];
    v8f acc;
#pragma unroll
    for (int r = 0; r < 8; ++r) acc[r] = bias;
#pragma unroll
    for (int kt = 0; kt < 6; ++kt) {
      v16h a  = frag_row(lds_x, mt * 16 + col, DIM, kt * 32);
      v16h bm = frag_swz(wqkv, jt * 6 + kt);
      acc = __builtin_amdgcn_wmma_f32_16x16x32_f16(false, a, false, bm,
                                                   (short)0, acc, false, false);
    }
    int which = jbase / DIM;        // 0=q 1=k 2=v (tiles never straddle)
    int cjb   = jbase - which * DIM;
#pragma unroll
    for (int r = 0; r < 8; ++r) {
      int n  = mt * 16 + r + radd;
      int cj = cjb + col;
      if (which == 2) lds_v[n * DIM + cj] = acc[r];               // v f32 for LePE
      else            lds_qk[which * (NTOK * DIM) + n * DIM + cj] = (_Float16)acc[r];
    }
  }
  __syncthreads();   // GEMM done; lds_x free for reuse

  // coalesced b128 store of q, k : dst layout (w, head, n, d)
  {
    u4* gq = (u4*)(qf + (size_t)w * NTOK * DIM);
    u4* gk = (u4*)(kf + (size_t)w * NTOK * DIM);
    for (int i4 = tid; i4 < NTOK * DIM / 8; i4 += 256) {
      int i = i4 * 8;
      int head = i >> 11;         // / (64*32)
      int n    = (i >> 5) & 63;
      int d0   = i & 31;          // multiple of 8
      gq[i4] = *(const u4*)(lds_qk + n * DIM + head * HD + d0);
      gk[i4] = *(const u4*)(lds_qk + NTOK * DIM + n * DIM + head * HD + d0);
    }
  }

  // LePE depthwise 3x3 in-window (zero pad); v' = v + conv(v).
  // Write v' transposed into padded LDS: vt[ch][n], stride VT_LD.
  for (int idx = tid; idx < NTOK * DIM; idx += 256) {
    int n = idx / DIM, ch = idx - n * DIM;
    int r = n >> 3, c = n & 7;
    float a = lepe_b[ch];
#pragma unroll
    for (int dr = 0; dr < 3; ++dr)
#pragma unroll
      for (int dc = 0; dc < 3; ++dc) {
        int rr = r + dr - 1, cc = c + dc - 1;
        if (rr >= 0 && rr < 8 && cc >= 0 && cc < 8)
          a += lds_v[((rr << 3) + cc) * DIM + ch] * lepe_w[(dr * 3 + dc) * DIM + ch];
      }
    float vp = lds_v[idx] + a;
    lds_x[ch * VT_LD + n] = (_Float16)vp;
  }
  __syncthreads();

  // coalesced b128 store of v' : dst layout (w, head, d, n)
  {
    u4* gv = (u4*)(vf + (size_t)w * NTOK * DIM);
    for (int i4 = tid; i4 < NTOK * DIM / 8; i4 += 256) {
      int i = i4 * 8;
      int ch = i >> 6;            // head*32 + d
      int n0 = i & 63;            // multiple of 8
      gv[i4] = *(const u4*)(lds_x + ch * VT_LD + n0);
    }
  }
}

// ------- Kernel 2: per (window, head) gated attention + softmax -------
__global__ __launch_bounds__(128) void k_attn(
    const _Float16* __restrict__ qf, const _Float16* __restrict__ kf,
    const _Float16* __restrict__ vf, const float* __restrict__ prev,
    float* __restrict__ attn_out, _Float16* __restrict__ aout) {
  __shared__ __align__(16) _Float16 lds_q[NTOK * HD];    // (n, d)
  __shared__ __align__(16) _Float16 lds_k[NTOK * HD];    // (n, d)
  __shared__ __align__(16) _Float16 lds_v[HD * NTOK];    // (d, n) transposed
  __shared__ __align__(16) _Float16 lds_o[NTOK * HD];    // O staging (n, d)
  __shared__ __align__(16) float    lds_s[NTOK * NTOK];
  __shared__ __align__(16) _Float16 lds_p[NTOK * NTOK];
  int wb = blockIdx.x;
  int w = wb / NH, h = wb - w * NH;
  size_t base  = ((size_t)w * NH + h) * NTOK * HD;
  size_t pbase = ((size_t)w * NH + h) * NTOK * NTOK;
  int tid = threadIdx.x;

  {
    const u4* gq = (const u4*)(qf + base);
    const u4* gk = (const u4*)(kf + base);
    const u4* gv = (const u4*)(vf + base);
    u4* sq = (u4*)lds_q; u4* sk = (u4*)lds_k; u4* sv = (u4*)lds_v;
    for (int i = tid; i < 256; i += 128) { sq[i] = gq[i]; sk[i] = gk[i]; sv[i] = gv[i]; }
  }
  __builtin_prefetch(prev + pbase + (size_t)tid * 32, 0, 0);
  __syncthreads();

  int lane = tid & 31, wave = tid >> 5;
  int col  = lane & 15;
  int radd = (lane & 16) ? 8 : 0;
  const float scale = 0.17677669529663687f;  // 32^-0.5

  // S = (q @ k^T) * scale * prev
  for (int t = wave; t < 16; t += 4) {
    int mt = t >> 2, jt = t & 3;
    v16h a  = frag_row(lds_q, mt * 16 + col, HD, 0);
    v16h bm = frag_row(lds_k, jt * 16 + col, HD, 0);
    v8f acc = {};
    acc = __builtin_amdgcn_wmma_f32_16x16x32_f16(false, a, false, bm,
                                                 (short)0, acc, false, false);
#pragma unroll
    for (int r = 0; r < 8; ++r) {
      int qi = mt * 16 + r + radd;
      int ki = jt * 16 + col;
      float g = __builtin_nontemporal_load(prev + pbase + (size_t)qi * NTOK + ki);
      lds_s[qi * NTOK + ki] = acc[r] * scale * g;
    }
  }
  __syncthreads();

  // row softmax (normalized values left in lds_s)
  if (tid < NTOK) {
    f4* rowp = (f4*)(lds_s + tid * NTOK);
    float m = -3.0e38f;
#pragma unroll
    for (int i = 0; i < 16; ++i) {
      f4 v = rowp[i];
      m = fmaxf(m, fmaxf(fmaxf(v.x, v.y), fmaxf(v.z, v.w)));
    }
    float s = 0.f;
#pragma unroll
    for (int i = 0; i < 16; ++i) {
      f4 v = rowp[i];
      v.x = __expf(v.x - m); v.y = __expf(v.y - m);
      v.z = __expf(v.z - m); v.w = __expf(v.w - m);
      s += v.x + v.y + v.z + v.w;
      rowp[i] = v;
    }
    float inv = 1.f / s;
#pragma unroll
    for (int i = 0; i < 16; ++i) rowp[i] *= inv;
  }
  __syncthreads();

  // coalesced non-temporal attn-map emit + f16 P build
  {
    const f4* sp = (const f4*)lds_s;
    f4* gp = (f4*)(attn_out + pbase);
    for (int i = tid; i < NTOK * NTOK / 4; i += 128) {
      f4 v = sp[i];
      __builtin_nontemporal_store(v, gp + i);
      _Float16* d = lds_p + i * 4;
      d[0] = (_Float16)v.x; d[1] = (_Float16)v.y;
      d[2] = (_Float16)v.z; d[3] = (_Float16)v.w;
    }
  }
  __syncthreads();

  // O = P @ V -> LDS staging
  for (int t = wave; t < 8; t += 4) {
    int mt = t >> 1, jt = t & 1;
    v8f acc = {};
#pragma unroll
    for (int kt = 0; kt < 2; ++kt) {
      v16h a  = frag_row(lds_p, mt * 16 + col, NTOK, kt * 32);
      v16h bm = frag_row(lds_v, jt * 16 + col, NTOK, kt * 32);
      acc = __builtin_amdgcn_wmma_f32_16x16x32_f16(false, a, false, bm,
                                                   (short)0, acc, false, false);
    }
#pragma unroll
    for (int r = 0; r < 8; ++r)
      lds_o[(mt * 16 + r + radd) * HD + jt * 16 + col] = (_Float16)acc[r];
  }
  __syncthreads();

  // coalesced b128 store of O into aout (w, n, h*32 + d)
  for (int i4 = tid; i4 < NTOK * HD / 8; i4 += 128) {
    int i = i4 * 8;
    int n = i >> 5, d0 = i & 31;
    *(u4*)(aout + ((size_t)w * NTOK + n) * DIM + h * HD + d0) =
        *(const u4*)(lds_o + n * HD + d0);
  }
}

// ------- Kernel 3: output projection + window_reverse + roll scatter -------
__global__ __launch_bounds__(256) void k_proj(
    const _Float16* __restrict__ aout, const _Float16* __restrict__ wproj,
    const float* __restrict__ proj_b, float* __restrict__ out) {
  __shared__ __align__(16) _Float16 lds_a[NTOK * DIM];   // 24 KB
  __shared__ __align__(16) float    lds_of[NTOK * DIM];  // 48 KB output staging
  int w = blockIdx.x;
  int b = w >> 8, wh = (w >> 4) & 15, wwi = w & 15;
  int tid = threadIdx.x;
  {
    const u4* ga = (const u4*)(aout + (size_t)w * NTOK * DIM);
    u4* sa = (u4*)lds_a;
    for (int i = tid; i < NTOK * DIM / 8; i += 256) sa[i] = ga[i];
  }
  __syncthreads();

  int lane = tid & 31, wave = tid >> 5;
  int col  = lane & 15;
  int radd = (lane & 16) ? 8 : 0;

  // 64x192 @ 192x192 : (4 m-tiles) x (12 j-tiles), K = 192
  for (int t = wave; t < 48; t += 8) {
    int mt = t / 12, jt = t - mt * 12;
    int jbase = jt * 16;
    float bias = proj_b[jbase + col];
    v8f acc;
#pragma unroll
    for (int r = 0; r < 8; ++r) acc[r] = bias;
#pragma unroll
    for (int kt = 0; kt < 6; ++kt) {
      v16h a  = frag_row(lds_a, mt * 16 + col, DIM, kt * 32);
      v16h bm = frag_swz(wproj, jt * 6 + kt);
      acc = __builtin_amdgcn_wmma_f32_16x16x32_f16(false, a, false, bm,
                                                   (short)0, acc, false, false);
    }
#pragma unroll
    for (int r = 0; r < 8; ++r)
      lds_of[(mt * 16 + r + radd) * DIM + jbase + col] = acc[r];
  }
  __syncthreads();

  // coalesced non-temporal write-out with window_reverse + roll(+SHIFT)
  for (int i4 = tid; i4 < NTOK * DIM / 4; i4 += 256) {
    int f = i4 * 4;
    int n = f / DIM, ch = f - n * DIM;
    int rr = n >> 3, cc = n & 7;
    int oh = (wh * 8 + rr + SHIFT) & 127;
    int ow = (wwi * 8 + cc + SHIFT) & 127;
    f4 v = *(const f4*)(lds_of + f);
    __builtin_nontemporal_store(
        v, (f4*)(out + (((size_t)b * 128 + oh) * 128 + ow) * DIM + ch));
  }
}

extern "C" void kernel_launch(void* const* d_in, const int* in_sizes, int n_in,
                              void* d_out, int out_size, void* d_ws, size_t ws_size,
                              hipStream_t stream) {
  const float* x      = (const float*)d_in[0];
  const float* prev   = (const float*)d_in[1];
  const float* qkv_w  = (const float*)d_in[2];
  const float* qkv_b  = (const float*)d_in[3];
  const float* proj_w = (const float*)d_in[4];
  const float* proj_b = (const float*)d_in[5];
  const float* lepe_w = (const float*)d_in[6];
  const float* lepe_b = (const float*)d_in[7];

  // workspace layout (all f16)
  _Float16* wqkv  = (_Float16*)d_ws;          // 192*576 (swizzled frags)
  _Float16* wproj = wqkv + DIM * C3;          // 192*192 (swizzled frags)
  _Float16* qf    = wproj + DIM * DIM;        // WB*6*64*32  (n,d)
  const size_t per = (size_t)NWIN * NH * NTOK * HD;
  _Float16* kf    = qf + per;                 // (n,d)
  _Float16* vf    = kf + per;                 // (d,n) transposed
  _Float16* aout  = vf + per;                 // WB*64*192

  float* out  = (float*)d_out;                            // (4,128,128,192)
  float* attn = out + (size_t)NWIN * NTOK * DIM;          // (WB,6,64,64)

  k_cvt<<<(DIM * C3 + DIM * DIM + 255) / 256, 256, 0, stream>>>(qkv_w, proj_w, wqkv, wproj);
  k_qkv_lepe<<<NWIN, 256, 0, stream>>>(x, qkv_b, lepe_w, lepe_b, wqkv, qf, kf, vf);
  k_attn<<<NWIN * NH, 128, 0, stream>>>(qf, kf, vf, prev, attn, aout);
  k_proj<<<NWIN, 256, 0, stream>>>(aout, wproj, proj_b, out);
}
